// DLI_loss_2_6614249636377
// MI455X (gfx1250) — matrix operand order repaired and checked by reading, not compiled, not run on hardware
//
#include <hip/hip_runtime.h>

// Problem constants
#define B_ 16
#define S_ 4096
#define E_ 1024
#define T_ 32
#define H_ 512
#define G4H 2048          // 4*H
#define M2_ 480           // B*(T-2)
#define MR_ 512           // B*T gathered rows
#define NEGF (-1e30f)

typedef __attribute__((ext_vector_type(16))) __bf16 v16bf;
typedef __attribute__((ext_vector_type(8)))  __bf16 v8bf;   // 16 bytes -> b128 load
typedef __attribute__((ext_vector_type(8)))  float  v8f;

union V16 { v16bf v; v8bf h[2]; };

// Pack two f32 -> packed bf16 (round-to-nearest-even).
__device__ __forceinline__ unsigned pack2bf(float f0, float f1) {
    unsigned u0 = __float_as_uint(f0), u1 = __float_as_uint(f1);
    u0 = u0 + 0x7FFFu + ((u0 >> 16) & 1u);
    u1 = u1 + 0x7FFFu + ((u1 >> 16) & 1u);
    return (u0 >> 16) | (u1 & 0xFFFF0000u);
}

__device__ __forceinline__ float sigf(float x) { return 1.0f / (1.0f + __expf(-x)); }

// ------------------------------------------------ fp32 -> bf16 bulk convert
// Each thread converts 4 floats (one float4 load, two u32 stores).
__global__ void k_cvt(const float* __restrict__ src, unsigned* __restrict__ dst, int n4) {
    int i = blockIdx.x * blockDim.x + threadIdx.x;
    if (i < n4) {
        float4 v = ((const float4*)src)[i];
        dst[2 * i]     = pack2bf(v.x, v.y);
        dst[2 * i + 1] = pack2bf(v.z, v.w);
    }
}

// -------------------- gather: emit bf16 states AND t[r] = states_row . w_t
__global__ void k_gather_t(const float* __restrict__ enc, const int* __restrict__ ids,
                           const float* __restrict__ fcw,
                           unsigned* __restrict__ Sbf, float* __restrict__ t) {
    __shared__ float red[256];
    int r = blockIdx.x;                 // 0..511
    int b = r >> 5, tt = r & 31;
    int idx = ids[b * T_ + tt];
    const float* src = enc + ((size_t)b * S_ + (size_t)idx) * E_;
    const float* wt  = fcw + H_;
    unsigned* dst = Sbf + (size_t)r * (E_ / 2);
    int e0 = threadIdx.x * 4;           // 256 threads * 4 = 1024 = E_
    float4 v = *(const float4*)(src + e0);
    float p = v.x * wt[e0] + v.y * wt[e0 + 1] + v.z * wt[e0 + 2] + v.w * wt[e0 + 3];
    dst[e0 / 2]     = pack2bf(v.x, v.y);
    dst[e0 / 2 + 1] = pack2bf(v.z, v.w);
    red[threadIdx.x] = p;
    __syncthreads();
    for (int s = 128; s > 0; s >>= 1) {
        if (threadIdx.x < s) red[threadIdx.x] += red[threadIdx.x + s];
        __syncthreads();
    }
    if (threadIdx.x == 0) t[r] = red[0];
}

// --------------------------- G = states @ w_ih^T + bias   (bf16 WMMA)
// grid(32,4), block(256): 8 waves/block, each wave a 16x64 strip (4 WMMA tiles).
__global__ void k_gemm_wih(const __bf16* __restrict__ S,   // 512 x 1024 bf16
                           const __bf16* __restrict__ W,   // 2048 x 1024 bf16
                           const float* __restrict__ bih,
                           const float* __restrict__ bhh,
                           float* __restrict__ G) {        // 512 x 2048 f32
    const int lane = threadIdx.x & 31;
    const int wave = threadIdx.x >> 5;
    const int mtile  = blockIdx.x * 16;
    const int ntile0 = (blockIdx.y * 8 + wave) * 64;
    const int row = lane & 15;
    const int hi  = lane >> 4;

    const __bf16* arow = S + (size_t)(mtile + row) * E_;
    const __bf16* brow[4];
    #pragma unroll
    for (int i = 0; i < 4; ++i)
        brow[i] = W + (size_t)(ntile0 + i * 16 + row) * E_;   // B col n == W row n

    v8f acc[4] = {};
    for (int k = 0; k < E_; k += 32) {
        __builtin_prefetch(arow + k + 256, 0, 1);             // global_prefetch_b8
        V16 a;
        a.h[0] = *(const v8bf*)(arow + k + hi * 8);           // K-run [k+hi*8, +8)
        a.h[1] = *(const v8bf*)(arow + k + 16 + hi * 8);      // K-run [k+16+hi*8, +8)
        #pragma unroll
        for (int i = 0; i < 4; ++i) {
            V16 b;
            b.h[0] = *(const v8bf*)(brow[i] + k + hi * 8);    // K-run [k+hi*8, +8)
            b.h[1] = *(const v8bf*)(brow[i] + k + 16 + hi * 8);
            acc[i] = __builtin_amdgcn_wmma_f32_16x16x32_bf16(false, a.v, false, b.v,
                                                             (short)0, acc[i], false, false);
        }
    }
    const int mbase = mtile + 8 * hi;
    #pragma unroll
    for (int i = 0; i < 4; ++i) {
        const int col  = ntile0 + i * 16 + (lane & 15);
        const float bias = bih[col] + bhh[col];
        #pragma unroll
        for (int r = 0; r < 8; ++r)
            G[(size_t)(mbase + r) * G4H + col] = acc[i][r] + bias;
    }
}

// -------------------------- LSTM step 1 (c0 = 0): emits C1 f32 + H1 bf16
__global__ void k_lstm1(const float* __restrict__ G, unsigned* __restrict__ H1bf,
                        float* __restrict__ C1) {
    int m2 = blockIdx.x;                        // 0..479
    int b = m2 / 30, j = m2 % 30;
    const float* g = G + (size_t)(b * T_ + j) * G4H;
    int n = threadIdx.x * 2;                    // 256 threads * 2 = 512 = H_
    float i0 = sigf(g[n]),     i1 = sigf(g[n + 1]);
    float g0 = tanhf(g[2 * H_ + n]), g1 = tanhf(g[2 * H_ + n + 1]);
    float o0 = sigf(g[3 * H_ + n]), o1 = sigf(g[3 * H_ + n + 1]);
    float c0 = i0 * g0, c1 = i1 * g1;           // sig(f)*c_prev == 0
    float h0 = o0 * tanhf(c0), h1 = o1 * tanhf(c1);
    C1[(size_t)m2 * H_ + n]     = c0;
    C1[(size_t)m2 * H_ + n + 1] = c1;
    H1bf[((size_t)m2 * H_ + n) / 2] = pack2bf(h0, h1);
}

// ---------------- gates2 = G[x1 rows] + h1 @ w_hh^T   (bf16 WMMA)
// grid(30,4), block(256)
__global__ void k_gemm_whh(const __bf16* __restrict__ H1,  // 480 x 512 bf16
                           const __bf16* __restrict__ W,   // 2048 x 512 bf16
                           const float* __restrict__ G,    // 512 x 2048 (C init)
                           float* __restrict__ G2) {       // 480 x 2048
    const int lane = threadIdx.x & 31;
    const int wave = threadIdx.x >> 5;
    const int mtile  = blockIdx.x * 16;
    const int ntile0 = (blockIdx.y * 8 + wave) * 64;
    const int row = lane & 15;
    const int hi  = lane >> 4;

    const __bf16* arow = H1 + (size_t)(mtile + row) * H_;
    const __bf16* brow[4];
    #pragma unroll
    for (int i = 0; i < 4; ++i)
        brow[i] = W + (size_t)(ntile0 + i * 16 + row) * H_;

    const int mbase = mtile + 8 * hi;
    v8f acc[4];
    #pragma unroll
    for (int i = 0; i < 4; ++i) {
        const int col = ntile0 + i * 16 + (lane & 15);
        #pragma unroll
        for (int r = 0; r < 8; ++r) {
            int m2 = mbase + r;
            int grow = (m2 / 30) * T_ + (m2 % 30) + 1;        // x1 row of G
            acc[i][r] = G[(size_t)grow * G4H + col];
        }
    }
    for (int k = 0; k < H_; k += 32) {
        V16 a;
        a.h[0] = *(const v8bf*)(arow + k + hi * 8);
        a.h[1] = *(const v8bf*)(arow + k + 16 + hi * 8);
        #pragma unroll
        for (int i = 0; i < 4; ++i) {
            V16 b;
            b.h[0] = *(const v8bf*)(brow[i] + k + hi * 8);
            b.h[1] = *(const v8bf*)(brow[i] + k + 16 + hi * 8);
            acc[i] = __builtin_amdgcn_wmma_f32_16x16x32_bf16(false, a.v, false, b.v,
                                                             (short)0, acc[i], false, false);
        }
    }
    #pragma unroll
    for (int i = 0; i < 4; ++i) {
        const int col = ntile0 + i * 16 + (lane & 15);
        #pragma unroll
        for (int r = 0; r < 8; ++r)
            G2[(size_t)(mbase + r) * G4H + col] = acc[i][r];
    }
}

// ------------------------------------------------------- LSTM step 2
__global__ void k_lstm2(const float* __restrict__ G2, const float* __restrict__ C1,
                        float* __restrict__ H2) {
    int m2 = blockIdx.x;
    const float* g = G2 + (size_t)m2 * G4H;
    for (int n = threadIdx.x; n < H_; n += blockDim.x) {
        float ig = sigf(g[n]);
        float fg = sigf(g[H_ + n]);
        float gg = tanhf(g[2 * H_ + n]);
        float og = sigf(g[3 * H_ + n]);
        float c  = fg * C1[(size_t)m2 * H_ + n] + ig * gg;
        H2[(size_t)m2 * H_ + n] = og * tanhf(c);
    }
}

// ------------------------------------------ a[m2] = h2 . w_h + fc_b
__global__ void k_dots(const float* __restrict__ H2, const float* __restrict__ fcw,
                       const float* __restrict__ fcb, float* __restrict__ a) {
    __shared__ float red[256];
    int id = blockIdx.x;                 // 0..479
    const float* vec = H2 + (size_t)id * H_;
    float p = 0.f;
    for (int e = threadIdx.x; e < H_; e += blockDim.x) p += vec[e] * fcw[e];
    red[threadIdx.x] = p;
    __syncthreads();
    for (int s = 128; s > 0; s >>= 1) {
        if (threadIdx.x < s) red[threadIdx.x] += red[threadIdx.x + s];
        __syncthreads();
    }
    if (threadIdx.x == 0) a[id] = red[0] + fcb[0];
}

// --------------------------------------- masked logsumexp + mean (1 block)
__global__ void k_final(const float* __restrict__ a, const float* __restrict__ t,
                        float* __restrict__ out) {
    __shared__ float red[512];
    int idx = threadIdx.x;
    float part = 0.f;
    if (idx < M2_) {
        int b = idx / 30, j = idx % 30;
        float av = a[idx];
        const float* tb = t + b * T_;
        float lg[T_];
        float mx = NEGF;
        #pragma unroll
        for (int k = 0; k < T_; ++k) {
            float l = (k >= j + 2) ? (av + tb[k]) : NEGF;
            lg[k] = l;
            mx = fmaxf(mx, l);
        }
        float s = 0.f;
        #pragma unroll
        for (int k = 0; k < T_; ++k) s += __expf(lg[k] - mx);   // masked terms underflow
        float lse = mx + __logf(s);
        float correct = av + tb[j + 2];
        part = lse - correct;
    }
    red[idx] = part;
    __syncthreads();
    for (int s = 256; s > 0; s >>= 1) {
        if (idx < s) red[idx] += red[idx + s];
        __syncthreads();
    }
    if (idx == 0) out[0] = red[0] / (float)M2_;
}

extern "C" void kernel_launch(void* const* d_in, const int* in_sizes, int n_in,
                              void* d_out, int out_size, void* d_ws, size_t ws_size,
                              hipStream_t stream) {
    const float* enc = (const float*)d_in[0];   // (16,4096,1024)
    const int*   ids = (const int*)d_in[1];     // (16,32)
    const float* wih = (const float*)d_in[2];   // (2048,1024)
    const float* whh = (const float*)d_in[3];   // (2048,512)
    const float* bih = (const float*)d_in[4];   // (2048,)
    const float* bhh = (const float*)d_in[5];   // (2048,)
    const float* fcw = (const float*)d_in[6];   // (1536,)
    const float* fcb = (const float*)d_in[7];   // (1,)
    float* out = (float*)d_out;

    // Workspace layout (bytes), total ~17.9 MB, 256B-aligned regions
    unsigned char* ws = (unsigned char*)d_ws;
    size_t o = 0;
    unsigned* Sbf   = (unsigned*)(ws + o); o += (size_t)MR_ * E_ * 2;        // bf16 states
    unsigned* Wihbf = (unsigned*)(ws + o); o += (size_t)G4H * E_ * 2;        // bf16 w_ih
    unsigned* Whhbf = (unsigned*)(ws + o); o += (size_t)G4H * H_ * 2;        // bf16 w_hh
    float*    G     = (float*)   (ws + o); o += (size_t)MR_ * G4H * 4;
    float*    G2    = (float*)   (ws + o); o += (size_t)M2_ * G4H * 4;
    unsigned* H1bf  = (unsigned*)(ws + o); o += (size_t)M2_ * H_ * 2;
    float*    C1    = (float*)   (ws + o); o += (size_t)M2_ * H_ * 4;
    float*    H2    = (float*)   (ws + o); o += (size_t)M2_ * H_ * 4;
    float*    av    = (float*)   (ws + o); o += 512 * 4;
    float*    tv    = (float*)   (ws + o); o += 512 * 4;

    // Convert weights to bf16 (memory-bound, one pass)
    k_cvt<<<dim3((G4H * E_ / 4) / 256), dim3(256), 0, stream>>>(wih, Wihbf, G4H * E_ / 4);
    k_cvt<<<dim3((G4H * H_ / 4) / 256), dim3(256), 0, stream>>>(whh, Whhbf, G4H * H_ / 4);
    // Gather + bf16 convert + t dot (fp32)
    k_gather_t<<<dim3(MR_), dim3(256), 0, stream>>>(enc, ids, fcw, Sbf, tv);
    // LSTM pipeline on WMMA
    k_gemm_wih<<<dim3(MR_ / 16, 4), dim3(256), 0, stream>>>(
        (const __bf16*)Sbf, (const __bf16*)Wihbf, bih, bhh, G);
    k_lstm1<<<dim3(M2_), dim3(256), 0, stream>>>(G, H1bf, C1);
    k_gemm_whh<<<dim3(M2_ / 16, 4), dim3(256), 0, stream>>>(
        (const __bf16*)H1bf, (const __bf16*)Whhbf, G, G2);
    k_lstm2<<<dim3(M2_), dim3(256), 0, stream>>>(G2, C1, H2);
    // Head + reduction
    k_dots<<<dim3(M2_), dim3(256), 0, stream>>>(H2, fcw, fcb, av);
    k_final<<<dim3(1), dim3(512), 0, stream>>>(av, tv, out);
}
